// SocialModel_88536455840258
// MI455X (gfx1250) — compile-verified, matrix operand order checked by reading, not run on hardware
//
#include <hip/hip_runtime.h>
#include <hip/hip_bf16.h>
#include <stdint.h>

#define SEQ   256
#define NPED  4096
#define RNN   256
#define EMB   128
#define G2    32
#define KX    (EMB + RNN)     // 384 = concat(sp, h) contraction dim
#define GATES (4 * RNN)       // 1024 gate columns [i,f,g,o]

typedef __attribute__((ext_vector_type(16))) __bf16 v16bf;
typedef __attribute__((ext_vector_type(8)))  float  v8f;

union FragAB { v16bf v; uint4 q[2]; };

__device__ __forceinline__ unsigned short f2bf(float f) {
    unsigned u = __builtin_bit_cast(unsigned, f);
    u += 0x7FFFu + ((u >> 16) & 1u);   // round-to-nearest-even
    return (unsigned short)(u >> 16);
}

__device__ __forceinline__ float sigmoidf_(float x) {
    return 1.0f / (1.0f + __expf(-x));
}

// One-time prep: Wcat bf16 = [W_ih[:, :128] | W_hh]  (row-major [1024][384]),
// bcomb = b_ih + b_hh, wcol = W_ih[:,128], zero h-region of X and h/c state.
__global__ __launch_bounds__(256)
void prep_kernel(const float* __restrict__ W_ih, const float* __restrict__ W_hh,
                 const float* __restrict__ b_ih, const float* __restrict__ b_hh,
                 unsigned short* __restrict__ Wcat, float* __restrict__ bcomb,
                 float* __restrict__ wcol, unsigned short* __restrict__ X,
                 float* __restrict__ hc_out) {
    int tid = blockIdx.x * blockDim.x + threadIdx.x;
    int nt  = gridDim.x * blockDim.x;
    for (int i = tid; i < GATES * KX; i += nt) {
        int r = i / KX, k = i - r * KX;
        float w = (k < EMB) ? W_ih[r * (EMB + 1) + k] : W_hh[r * RNN + (k - EMB)];
        Wcat[i] = f2bf(w);
    }
    for (int r = tid; r < GATES; r += nt) {
        bcomb[r] = b_ih[r] + b_hh[r];
        wcol[r]  = W_ih[r * (EMB + 1) + EMB];
    }
    for (int i = tid; i < NPED * RNN; i += nt) {
        int p = i / RNN, n = i - p * RNN;
        X[p * KX + EMB + n] = 0;              // bf16 zero
    }
    for (int i = tid; i < 2 * NPED * RNN; i += nt) hc_out[i] = 0.0f;
}

// Per-step embeddings: sp[p][e] (bf16 into X), tn[p] (f32).
__global__ __launch_bounds__(256)
void embed_kernel(const float* __restrict__ frame,  // [NPED][5]
                  const float* __restrict__ grid,   // [NPED][G2]
                  const float* __restrict__ Ww, const float* __restrict__ bw,
                  const float* __restrict__ eb,
                  const float* __restrict__ Wt, const float* __restrict__ bt,
                  const float* __restrict__ et,
                  unsigned short* __restrict__ X, float* __restrict__ tn) {
    int tid = blockIdx.x * blockDim.x + threadIdx.x;   // NPED*EMB threads
    int p = tid >> 7;
    int e = tid & 127;
    if (p >= NPED) return;
    const float* fr = frame + p * 5;
    float s = bw[e] + eb[e]
            + Ww[e * 4 + 0] * fr[1] + Ww[e * 4 + 1] * fr[2]
            + Ww[e * 4 + 2] * fr[3] + Ww[e * 4 + 3] * fr[4];
    X[p * KX + e] = f2bf(fmaxf(s, 0.0f));
    if (e == 0) {
        float t = bt[0] + et[0];
        const float* g = grid + p * G2;
        #pragma unroll
        for (int j = 0; j < G2; ++j) t += Wt[j] * g[j];
        tn[p] = fmaxf(t, 0.0f);
    }
}

// Fused LSTM step: gates GEMM via v_wmma_f32_16x16x32_bf16 + activations +
// masked state update. Block tile: 64 peds x 64 rnn cols x all 4 gates.
__global__ __launch_bounds__(256)
void lstm_step_kernel(const unsigned short* __restrict__ X,    // [NPED][KX] bf16
                      const unsigned short* __restrict__ Wcat, // [GATES][KX] bf16
                      const float* __restrict__ bcomb,         // [GATES]
                      const float* __restrict__ wcol,          // [GATES]
                      const float* __restrict__ tn,            // [NPED]
                      const float* __restrict__ frame,         // [NPED][5]
                      float* __restrict__ h_out,               // [NPED][RNN]
                      float* __restrict__ c_out,               // [NPED][RNN]
                      unsigned short* __restrict__ Xh) {       // == X (h bf16 region)
    __shared__ float lds[4 * 64 * 64];   // [gate][m_local][n_local]

    const int n0 = blockIdx.x * 64;      // rnn column tile
    const int m0 = blockIdx.y * 64;      // ped row tile

    const int lane = threadIdx.x & 31;
    const int wave = threadIdx.x >> 5;   // 8 waves
    const int gate = wave >> 1;          // 0..3 = i,f,g,o
    const int mh   = wave & 1;           // 32-row half

    const int l15  = lane & 15;
    const int half = lane >> 4;

    v8f acc[2][4] = {};                  // 2 M-subtiles x 4 N-subtiles

    for (int ko = 0; ko < KX; ko += 32) {
        FragAB a[2], b[4];
        #pragma unroll
        for (int mi = 0; mi < 2; ++mi) {
            int row = m0 + mh * 32 + mi * 16 + l15;          // lane = M row
            const unsigned short* ap = X + row * KX + ko + half * 8;
            a[mi].q[0] = *(const uint4*)(ap);                // K = ko+half*8 .. +8
            a[mi].q[1] = *(const uint4*)(ap + 16);           // K = ko+16+half*8 .. +8
        }
        #pragma unroll
        for (int ni = 0; ni < 4; ++ni) {
            int gcol = gate * RNN + n0 + ni * 16 + l15;      // lane = N col
            const unsigned short* bp = Wcat + gcol * KX + ko + half * 8;
            b[ni].q[0] = *(const uint4*)(bp);
            b[ni].q[1] = *(const uint4*)(bp + 16);
        }
        #pragma unroll
        for (int mi = 0; mi < 2; ++mi)
            #pragma unroll
            for (int ni = 0; ni < 4; ++ni)
                acc[mi][ni] = __builtin_amdgcn_wmma_f32_16x16x32_bf16(
                    false, a[mi].v, false, b[ni].v, (short)0, acc[mi][ni],
                    false, false);
    }

    // C/D layout: element e -> M = e + 8*half ; lane&15 -> N
    float* lg = lds + gate * 64 * 64;
    #pragma unroll
    for (int mi = 0; mi < 2; ++mi)
        #pragma unroll
        for (int ni = 0; ni < 4; ++ni)
            #pragma unroll
            for (int e = 0; e < 8; ++e) {
                int ml = mh * 32 + mi * 16 + e + 8 * half;
                int nl = ni * 16 + l15;
                lg[ml * 64 + nl] = acc[mi][ni][e];
            }
    __syncthreads();

    // Fused elementwise: bias + rank-1 tn term + LSTM activations + mask.
    for (int idx = threadIdx.x; idx < 64 * 64; idx += 256) {
        int r = idx >> 6;
        int n = idx & 63;
        int p    = m0 + r;
        int ncol = n0 + n;
        if (frame[p * 5] == 0.0f) continue;   // inactive ped: keep state
        float tv = tn[p];
        float gi = lds[0 * 4096 + idx] + bcomb[0 * RNN + ncol] + tv * wcol[0 * RNN + ncol];
        float gf = lds[1 * 4096 + idx] + bcomb[1 * RNN + ncol] + tv * wcol[1 * RNN + ncol];
        float gg = lds[2 * 4096 + idx] + bcomb[2 * RNN + ncol] + tv * wcol[2 * RNN + ncol];
        float go = lds[3 * 4096 + idx] + bcomb[3 * RNN + ncol] + tv * wcol[3 * RNN + ncol];
        float c_old = c_out[p * RNN + ncol];
        float cn = sigmoidf_(gf) * c_old + sigmoidf_(gi) * tanhf(gg);
        float hn = sigmoidf_(go) * tanhf(cn);
        c_out[p * RNN + ncol] = cn;
        h_out[p * RNN + ncol] = hn;
        Xh[p * KX + EMB + ncol] = f2bf(hn);   // bf16 h for next step's GEMM
    }
}

extern "C" void kernel_launch(void* const* d_in, const int* in_sizes, int n_in,
                              void* d_out, int out_size, void* d_ws, size_t ws_size,
                              hipStream_t stream) {
    const float* input_data = (const float*)d_in[0];
    const float* grid_data  = (const float*)d_in[1];
    const float* Ww   = (const float*)d_in[2];
    const float* bw   = (const float*)d_in[3];
    const float* eb   = (const float*)d_in[4];
    const float* Wt   = (const float*)d_in[5];
    const float* bt   = (const float*)d_in[6];
    const float* et   = (const float*)d_in[7];
    const float* W_ih = (const float*)d_in[8];
    const float* W_hh = (const float*)d_in[9];
    const float* b_ih = (const float*)d_in[10];
    const float* b_hh = (const float*)d_in[11];
    // d_in[12..14] (Wo, bo, ob) unused: reference returns only (h, c)

    char* ws = (char*)d_ws;
    size_t off = 0;
    auto walloc = [&](size_t bytes) {
        char* p = ws + off;
        off = (off + bytes + 255) & ~(size_t)255;
        return p;
    };
    unsigned short* X    = (unsigned short*)walloc((size_t)NPED * KX * 2);
    unsigned short* Wcat = (unsigned short*)walloc((size_t)GATES * KX * 2);
    float* bcomb = (float*)walloc(GATES * 4);
    float* wcol  = (float*)walloc(GATES * 4);
    float* tn    = (float*)walloc(NPED * 4);
    (void)ws_size; (void)in_sizes; (void)n_in; (void)out_size;

    float* h_out = (float*)d_out;              // [NPED][RNN]
    float* c_out = h_out + (size_t)NPED * RNN; // [NPED][RNN]

    prep_kernel<<<512, 256, 0, stream>>>(W_ih, W_hh, b_ih, b_hh,
                                         Wcat, bcomb, wcol, X, h_out);

    dim3 lgrid(RNN / 64, NPED / 64);           // (4, 64)
    for (int t = 0; t < SEQ; ++t) {
        const float* frame = input_data + (size_t)t * NPED * 5;
        const float* grid  = grid_data  + (size_t)t * NPED * G2;
        embed_kernel<<<(NPED * EMB) / 256, 256, 0, stream>>>(
            frame, grid, Ww, bw, eb, Wt, bt, et, X, tn);
        lstm_step_kernel<<<lgrid, 256, 0, stream>>>(
            X, Wcat, bcomb, wcol, tn, frame, h_out, c_out, X);
    }
}
